// LinearAdaMoleLayer_3977139716765
// MI455X (gfx1250) — matrix operand order repaired and checked by reading, not compiled, not run on hardware
//
#include <hip/hip_runtime.h>
#include <hip/hip_bf16.h>

// Problem constants (fixed by the reference)
#define TOK   8192      // B*S
#define DD    4096
#define OO    4096
#define EE    8
#define RR    16
#define KLORA 128       // E*R
#define SCALING 2.0f
#define MAXTHR  0.125f

#define KS    32        // K step per WMMA stage (bf16 k=32)
#define LDST  40        // LDS row stride in bf16 elements (32 + 8 pad, keeps 16B alignment)

typedef __attribute__((ext_vector_type(16))) __bf16 v16bf;
typedef __attribute__((ext_vector_type(8)))  __bf16 v8bf;
typedef __attribute__((ext_vector_type(8)))  float  v8f;

static __device__ __forceinline__ v16bf cat16(v8bf lo, v8bf hi) {
  return __builtin_shufflevector(lo, hi, 0,1,2,3,4,5,6,7,8,9,10,11,12,13,14,15);
}
static __device__ __forceinline__ v8bf cvt8(float4 a, float4 b) {
  v8bf r;
  r[0] = (__bf16)a.x; r[1] = (__bf16)a.y; r[2] = (__bf16)a.z; r[3] = (__bf16)a.w;
  r[4] = (__bf16)b.x; r[5] = (__bf16)b.y; r[6] = (__bf16)b.z; r[7] = (__bf16)b.w;
  return r;
}

// ---------------------------------------------------------------------------
// Kernel 1: gating. One wave32 per token: 8 gate logits + 1 threshold logit,
// wave-reduce, softmax + sigmoid threshold + renormalize -> w_s[N,8].
// ---------------------------------------------------------------------------
__global__ __launch_bounds__(256)
void gating_kernel(const float* __restrict__ x, const float* __restrict__ Wg,
                   const float* __restrict__ Wthr, const float* __restrict__ bthr,
                   float* __restrict__ w_s)
{
  const int lane = threadIdx.x & 31;
  const int n = blockIdx.x * 8 + (threadIdx.x >> 5);
  const float* xr = x + (size_t)n * DD;

  float acc[9];
  #pragma unroll
  for (int i = 0; i < 9; ++i) acc[i] = 0.f;

  for (int d = lane * 4; d < DD; d += 128) {
    float4 xv = *(const float4*)(xr + d);
    #pragma unroll
    for (int e = 0; e < EE; ++e) {
      float4 wv = *(const float4*)(Wg + (size_t)e * DD + d);
      acc[e] += xv.x*wv.x + xv.y*wv.y + xv.z*wv.z + xv.w*wv.w;
    }
    float4 tv = *(const float4*)(Wthr + d);
    acc[8] += xv.x*tv.x + xv.y*tv.y + xv.z*tv.z + xv.w*tv.w;
  }

  #pragma unroll
  for (int i = 0; i < 9; ++i) {
    #pragma unroll
    for (int off = 16; off > 0; off >>= 1)
      acc[i] += __shfl_down(acc[i], off, 32);
  }

  if (lane == 0) {
    float mx = acc[0];
    #pragma unroll
    for (int e = 1; e < EE; ++e) mx = fmaxf(mx, acc[e]);
    float p[EE]; float s = 0.f;
    #pragma unroll
    for (int e = 0; e < EE; ++e) { p[e] = __expf(acc[e] - mx); s += p[e]; }
    const float inv = 1.f / s;
    const float thr = MAXTHR / (1.f + __expf(-(acc[8] + bthr[0])));
    float w[EE]; float ws = 0.f;
    #pragma unroll
    for (int e = 0; e < EE; ++e) {
      float a = p[e] * inv - thr;
      w[e] = a > 0.f ? a : 0.f;
      ws += w[e];
    }
    if (ws == 0.f) ws = 1.f;
    const float iw = 1.f / ws;
    #pragma unroll
    for (int e = 0; e < EE; ++e) w_s[(size_t)n * EE + e] = w[e] * iw;
  }
}

// ---------------------------------------------------------------------------
// Kernel 2: repack B_w[E,O,R]*SCALING -> Bt[O, E*R] bf16 (K-major like W_base)
// ---------------------------------------------------------------------------
__global__ __launch_bounds__(256)
void pack_bt_kernel(const float* __restrict__ Bw, __bf16* __restrict__ Bt)
{
  const int i = blockIdx.x * 256 + threadIdx.x;   // [0, OO*KLORA)
  const int c = i & (KLORA - 1);
  const int o = i >> 7;
  const int e = c >> 4, r = c & 15;
  Bt[i] = (__bf16)(Bw[((size_t)e * OO + o) * RR + r] * SCALING);
}

// ---------------------------------------------------------------------------
// Kernel 3: Hw[n, e*16+r] = bf16( w_s[n,e] * sum_d x[n,d] * A_w[e,r,d] )
// WMMA GEMM: M=8192 (64 blocks of 128), N=128 (single tile), K=4096.
// ---------------------------------------------------------------------------
__global__ __launch_bounds__(256)
void lora_h_kernel(const float* __restrict__ x, const float* __restrict__ Aw,
                   const float* __restrict__ w_s, __bf16* __restrict__ Hw)
{
  __shared__ __bf16 As[2][128 * LDST];
  __shared__ __bf16 Bs[2][128 * LDST];

  const int tid  = threadIdx.x;
  const int lane = tid & 31;
  const int wave = tid >> 5;
  const int row0 = blockIdx.x * 128;

  const int l16 = lane & 15;
  const int lh  = lane >> 4;
  const int wm  = (wave >> 1) * 32;   // 0,32,64,96
  const int wn  = (wave & 1) * 64;    // 0,64

  const int lr = tid & 127;
  const int lc = (tid >> 7) << 4;     // 0 or 16

  const float* ag = x  + (size_t)(row0 + lr) * DD + lc;   // token rows
  const float* bg = Aw + (size_t)lr * DD + lc;            // A_w flat [128, D]

  v8f acc[2][4];
  #pragma unroll
  for (int im = 0; im < 2; ++im)
    #pragma unroll
    for (int jn = 0; jn < 4; ++jn) acc[im][jn] = (v8f)(0.f);

  float4 ra[4], rb[4];
  auto loadG = [&](int ks) {
    const float* pa = ag + ks * KS;
    ra[0] = *(const float4*)(pa + 0);  ra[1] = *(const float4*)(pa + 4);
    ra[2] = *(const float4*)(pa + 8);  ra[3] = *(const float4*)(pa + 12);
    const float* pb = bg + ks * KS;
    rb[0] = *(const float4*)(pb + 0);  rb[1] = *(const float4*)(pb + 4);
    rb[2] = *(const float4*)(pb + 8);  rb[3] = *(const float4*)(pb + 12);
  };
  auto storeL = [&](int buf) {
    __bf16* da = &As[buf][lr * LDST + lc];
    *(v8bf*)(da)     = cvt8(ra[0], ra[1]);
    *(v8bf*)(da + 8) = cvt8(ra[2], ra[3]);
    __bf16* db = &Bs[buf][lr * LDST + lc];
    *(v8bf*)(db)     = cvt8(rb[0], rb[1]);
    *(v8bf*)(db + 8) = cvt8(rb[2], rb[3]);
  };
  auto compute_step = [&](int buf) {
    v16bf af[2], bf[4];
    #pragma unroll
    for (int im = 0; im < 2; ++im) {
      const __bf16* p = &As[buf][(wm + im*16 + l16) * LDST + lh * 8];
      af[im] = cat16(*(const v8bf*)p, *(const v8bf*)(p + 16));
    }
    #pragma unroll
    for (int jn = 0; jn < 4; ++jn) {
      const __bf16* p = &Bs[buf][(wn + jn*16 + l16) * LDST + lh * 8];
      bf[jn] = cat16(*(const v8bf*)p, *(const v8bf*)(p + 16));
    }
    #pragma unroll
    for (int im = 0; im < 2; ++im)
      #pragma unroll
      for (int jn = 0; jn < 4; ++jn)
        acc[im][jn] = __builtin_amdgcn_wmma_f32_16x16x32_bf16(
            false, af[im], false, bf[jn], (short)0, acc[im][jn], false, false);
  };

  loadG(0); storeL(0);
  __syncthreads();
  #pragma unroll 1
  for (int ks = 0; ks < DD / KS; ++ks) {
    if (ks + 1 < DD / KS) loadG(ks + 1);
    compute_step(ks & 1);
    if (ks + 1 < DD / KS) storeL((ks + 1) & 1);
    __syncthreads();
  }

  // Epilogue: scale by per-token expert weight, store bf16.
  #pragma unroll
  for (int jn = 0; jn < 4; ++jn) {
    const int c = wn + jn * 16 + l16;
    const int e = c >> 4;
    #pragma unroll
    for (int im = 0; im < 2; ++im) {
      const int mbase = row0 + wm + im * 16 + lh * 8;
      #pragma unroll
      for (int v = 0; v < 8; ++v) {
        const int row = mbase + v;
        const float wv = w_s[(size_t)row * EE + e];
        Hw[(size_t)row * KLORA + c] = (__bf16)(acc[im][jn][v] * wv);
      }
    }
  }
}

// ---------------------------------------------------------------------------
// Kernel 4: out[n,o] = b_base[o] + sum_{d<4096} x*Wb + sum_{c<128} Hw*Bt
// 128x128 tile, 8 waves, augmented-K WMMA GEMM with double-buffered LDS.
// ---------------------------------------------------------------------------
__global__ __launch_bounds__(256)
void main_gemm_kernel(const float* __restrict__ x, const float* __restrict__ Wb,
                      const float* __restrict__ bb, const __bf16* __restrict__ Hw,
                      const __bf16* __restrict__ Bt, float* __restrict__ out)
{
  __shared__ __bf16 As[2][128 * LDST];
  __shared__ __bf16 Bs[2][128 * LDST];

  const int tid  = threadIdx.x;
  const int lane = tid & 31;
  const int wave = tid >> 5;
  const int row0 = blockIdx.x * 128;
  const int col0 = blockIdx.y * 128;

  const int l16 = lane & 15;
  const int lh  = lane >> 4;
  const int wm  = (wave >> 1) * 32;
  const int wn  = (wave & 1) * 64;

  const int lr = tid & 127;
  const int lc = (tid >> 7) << 4;

  const float* ag = x  + (size_t)(row0 + lr) * DD + lc;
  const float* bg = Wb + (size_t)(col0 + lr) * DD + lc;

  v8f acc[2][4];
  #pragma unroll
  for (int im = 0; im < 2; ++im)
    #pragma unroll
    for (int jn = 0; jn < 4; ++jn) acc[im][jn] = (v8f)(0.f);

  float4 ra[4], rb[4];
  auto loadG = [&](int ks) {
    const float* pa = ag + ks * KS;
    ra[0] = *(const float4*)(pa + 0);  ra[1] = *(const float4*)(pa + 4);
    ra[2] = *(const float4*)(pa + 8);  ra[3] = *(const float4*)(pa + 12);
    const float* pb = bg + ks * KS;
    rb[0] = *(const float4*)(pb + 0);  rb[1] = *(const float4*)(pb + 4);
    rb[2] = *(const float4*)(pb + 8);  rb[3] = *(const float4*)(pb + 12);
    if (ks + 8 < DD / KS) {                  // gfx1250 global_prefetch_b8
      __builtin_prefetch(ag + (ks + 8) * KS, 0, 0);
      __builtin_prefetch(bg + (ks + 8) * KS, 0, 0);
    }
  };
  auto storeL = [&](int buf) {
    __bf16* da = &As[buf][lr * LDST + lc];
    *(v8bf*)(da)     = cvt8(ra[0], ra[1]);
    *(v8bf*)(da + 8) = cvt8(ra[2], ra[3]);
    __bf16* db = &Bs[buf][lr * LDST + lc];
    *(v8bf*)(db)     = cvt8(rb[0], rb[1]);
    *(v8bf*)(db + 8) = cvt8(rb[2], rb[3]);
  };
  auto compute_step = [&](int buf) {
    v16bf af[2], bf[4];
    #pragma unroll
    for (int im = 0; im < 2; ++im) {
      const __bf16* p = &As[buf][(wm + im*16 + l16) * LDST + lh * 8];
      af[im] = cat16(*(const v8bf*)p, *(const v8bf*)(p + 16));
    }
    #pragma unroll
    for (int jn = 0; jn < 4; ++jn) {
      const __bf16* p = &Bs[buf][(wn + jn*16 + l16) * LDST + lh * 8];
      bf[jn] = cat16(*(const v8bf*)p, *(const v8bf*)(p + 16));
    }
    #pragma unroll
    for (int im = 0; im < 2; ++im)
      #pragma unroll
      for (int jn = 0; jn < 4; ++jn)
        acc[im][jn] = __builtin_amdgcn_wmma_f32_16x16x32_bf16(
            false, af[im], false, bf[jn], (short)0, acc[im][jn], false, false);
  };

  // Phase 1: K over D (f32 -> bf16 convert, double buffered)
  loadG(0); storeL(0);
  __syncthreads();
  #pragma unroll 1
  for (int ks = 0; ks < DD / KS; ++ks) {
    if (ks + 1 < DD / KS) loadG(ks + 1);
    compute_step(ks & 1);
    if (ks + 1 < DD / KS) storeL((ks + 1) & 1);
    __syncthreads();
  }

  // Phase 2: augmented K over the 128 LoRA dims (already bf16)
  const __bf16* hg = Hw + (size_t)(row0 + lr) * KLORA + lc;
  const __bf16* tg = Bt + (size_t)(col0 + lr) * KLORA + lc;
  #pragma unroll 1
  for (int ks = 0; ks < KLORA / KS; ++ks) {
    v8bf h0 = *(const v8bf*)(hg + ks * KS);
    v8bf h1 = *(const v8bf*)(hg + ks * KS + 8);
    v8bf t0 = *(const v8bf*)(tg + ks * KS);
    v8bf t1 = *(const v8bf*)(tg + ks * KS + 8);
    __bf16* da = &As[0][lr * LDST + lc];
    *(v8bf*)(da)     = h0; *(v8bf*)(da + 8) = h1;
    __bf16* db = &Bs[0][lr * LDST + lc];
    *(v8bf*)(db)     = t0; *(v8bf*)(db + 8) = t1;
    __syncthreads();
    compute_step(0);
    __syncthreads();
  }

  // Epilogue: add bias, single f32 store of the fused result.
  #pragma unroll
  for (int jn = 0; jn < 4; ++jn) {
    const int c = col0 + wn + jn * 16 + l16;
    const float bias = bb[c];
    #pragma unroll
    for (int im = 0; im < 2; ++im) {
      const int mbase = row0 + wm + im * 16 + lh * 8;
      #pragma unroll
      for (int v = 0; v < 8; ++v)
        out[(size_t)(mbase + v) * OO + c] = acc[im][jn][v] + bias;
    }
  }
}

// ---------------------------------------------------------------------------
extern "C" void kernel_launch(void* const* d_in, const int* in_sizes, int n_in,
                              void* d_out, int out_size, void* d_ws, size_t ws_size,
                              hipStream_t stream) {
  const float* x    = (const float*)d_in[0];
  const float* Wb   = (const float*)d_in[1];
  const float* bb   = (const float*)d_in[2];
  const float* Wg   = (const float*)d_in[3];
  const float* Wthr = (const float*)d_in[4];
  const float* bthr = (const float*)d_in[5];
  const float* Aw   = (const float*)d_in[6];
  const float* Bw   = (const float*)d_in[7];
  float* out = (float*)d_out;

  // Workspace layout (~3.3 MB): w_s[N,8] f32 | Hw[N,128] bf16 | Bt[O,128] bf16
  float*  w_s = (float*)d_ws;
  __bf16* Hw  = (__bf16*)((char*)d_ws + (size_t)TOK * EE * sizeof(float));
  __bf16* Bt  = (__bf16*)((char*)Hw + (size_t)TOK * KLORA * sizeof(__bf16));

  gating_kernel<<<TOK / 8, 256, 0, stream>>>(x, Wg, Wthr, bthr, w_s);
  pack_bt_kernel<<<(OO * KLORA) / 256, 256, 0, stream>>>(Bw, Bt);
  lora_h_kernel<<<TOK / 128, 256, 0, stream>>>(x, Aw, w_s, Hw);
  main_gemm_kernel<<<dim3(TOK / 128, OO / 128), 256, 0, stream>>>(x, Wb, bb, Hw, Bt, out);
}
